// RecurrentAttention_8134668059214
// MI455X (gfx1250) — compile-verified, matrix operand order checked by reading, not run on hardware
//
#include <hip/hip_runtime.h>

#define DIMC 1024
#define HEADS 16
#define HEAD_DIM 64
#define STATE 16
#define BB 4
#define TT 1024
#define NROWS (BB*TT)

typedef __attribute__((ext_vector_type(16))) _Float16 v16h;
typedef __attribute__((ext_vector_type(8)))  _Float16 v8h;
typedef __attribute__((ext_vector_type(8)))  float    v8f;

// D = A(16x32 f16) * B(32x16 f16) + C(f32)
__device__ __forceinline__ v8f wmma_f16(v16h a, v16h b, v8f c) {
  return __builtin_amdgcn_wmma_f32_16x16x32_f16(
      /*neg_a=*/false, a, /*neg_b=*/false, b,
      /*c_mod=*/(short)0, c, /*reuse_a=*/false, /*reuse_b=*/false);
}

// ISA 7.12.2 16-bit A/B fragment layout: lane<16 -> K {0..7, 16..23},
// lane>=16 -> K {8..15, 24..31}. koff = (lane>>4)*8. Two 16-byte loads.
__device__ __forceinline__ v16h load_frag(const _Float16* __restrict__ row, int koff) {
  v8h lo = *(const v8h*)(row + koff);
  v8h hi = *(const v8h*)(row + koff + 16);
  v16h r;
#pragma unroll
  for (int i = 0; i < 8; ++i) { r[i] = lo[i]; r[8 + i] = hi[i]; }
  return r;
}

// Branchless async stage of a 64x32 f16 tile (4 KB) into LDS.
// 256 16-byte chunks, exactly 2 per thread (128 threads). ASYNCcnt-tracked.
__device__ __forceinline__ void stage_tile(
    const _Float16* __restrict__ W, int n0, int k0, int K,
    _Float16* buf, int tid) {
#pragma unroll
  for (int q = 0; q < 2; ++q) {
    int c   = tid + q * 128;
    int row = c >> 2;
    int kc  = (c & 3) << 3;
    const _Float16* src = W + (size_t)(n0 + row) * K + k0 + kc;
    unsigned lds_addr = (unsigned)(uintptr_t)(buf + row * 32 + kc);
    unsigned long long gaddr = (unsigned long long)(uintptr_t)src;
    asm volatile("global_load_async_to_lds_b128 %0, %1, off"
                 :: "v"(lds_addr), "v"(gaddr) : "memory");
  }
}

// ---------------- elementwise f32 -> f16 ----------------
__global__ void cvt_f32_f16(const float* __restrict__ src, _Float16* __restrict__ dst, int n) {
  int i = blockIdx.x * blockDim.x + threadIdx.x;
  if (i < n) dst[i] = (_Float16)src[i];
}

// ---------------- wide NT WMMA GEMM, double-buffered async LDS staging -----
// 4 waves / 128 threads; block computes 64(M) x 64(N). While the matrix
// pipes consume tile k from one LDS buffer, the async engine streams tile
// k+1 into the other (s_wait_asynccnt 0x2: in-order completion guarantees
// the current tile landed with the next still in flight).
__global__ __launch_bounds__(128) void wmma_gemm_lds(
    const _Float16* __restrict__ A, const _Float16* __restrict__ W,
    const float* __restrict__ bias, float* __restrict__ Y,
    int M, int N, int K)
{
  __shared__ __align__(16) _Float16 Btile[2][64 * 32];   // 2 x 4 KB

  int tid  = threadIdx.x;
  int wave = tid >> 5;
  int lane = tid & 31;
  int m    = lane & 15;
  int half = lane >> 4;
  int koff = half << 3;

  int m0 = blockIdx.x * 64 + wave * 16;   // this wave's 16 rows
  int n0 = blockIdx.y * 64;               // block's 64 cols

  const _Float16* arow = A + (size_t)(m0 + m) * K;
  v8f acc[4] = {};

  int nk = K >> 5;
  stage_tile(W, n0, 0, K, Btile[0], tid);   // prologue: tile 0 in flight

  for (int it = 0; it < nk; ++it) {
    int cur = it & 1;
    __syncthreads();   // all waves done reading Btile[cur^1] (prev iter)
    if (it + 1 < nk) {
      stage_tile(W, n0, (it + 1) << 5, K, Btile[cur ^ 1], tid);
      asm volatile("s_wait_asynccnt 0x2" ::: "memory");  // tile[cur] landed
    } else {
      asm volatile("s_wait_asynccnt 0x0" ::: "memory");
    }
    __syncthreads();   // tile[cur] visible to all waves

    v16h a = load_frag(arow + (it << 5), koff);
    __builtin_prefetch(arow + (it << 5) + 256, 0, 0);    // global_prefetch_b8

    v16h bf[4];
#pragma unroll
    for (int s = 0; s < 4; ++s) {
      const _Float16* bp = Btile[cur] + (s * 16 + m) * 32;
      v8h lo = *(const v8h*)(bp + koff);
      v8h hi = *(const v8h*)(bp + koff + 16);
#pragma unroll
      for (int i = 0; i < 8; ++i) { bf[s][i] = lo[i]; bf[s][8 + i] = hi[i]; }
    }
#pragma unroll
    for (int s = 0; s < 4; ++s)
      acc[s] = wmma_f16(a, bf[s], acc[s]);
  }

#pragma unroll
  for (int s = 0; s < 4; ++s) {
#pragma unroll
    for (int r = 0; r < 8; ++r) {
      int gm = m0 + r + 8 * half;
      int gn = n0 + s * 16 + m;
      float v = acc[s][r];
      if (bias) v += bias[gn];
      Y[(size_t)gm * N + gn] = v;
    }
  }
}

// ---------------- narrow NT WMMA GEMM (N = 16): one wave per 16x16 tile ----
__global__ __launch_bounds__(32) void wmma_gemm_narrow(
    const _Float16* __restrict__ A, const _Float16* __restrict__ W,
    float* __restrict__ Y, int M, int N, int K)
{
  int tile_m = blockIdx.x;
  int lane = threadIdx.x;
  int m    = lane & 15;
  int half = lane >> 4;
  int koff = half << 3;

  const _Float16* arow = A + (size_t)(tile_m * 16 + m) * K;
  const _Float16* wrow = W + (size_t)m * K;
  v8f acc = {};

  for (int k0 = 0; k0 < K; k0 += 32) {
    v16h a = load_frag(arow + k0, koff);
    v16h b = load_frag(wrow + k0, koff);
    acc = wmma_f16(a, b, acc);
  }

#pragma unroll
  for (int r = 0; r < 8; ++r)
    Y[(size_t)(tile_m * 16 + r + 8 * half) * N + m] = acc[r];
}

// ---------------- sequential selective scan ----------------
__global__ void scan_kernel(
    const float* __restrict__ xb,     // [B,T,DIM] x_base
    const float* __restrict__ delta,  // [B,T,DIM]
    const float* __restrict__ Bm,     // [B,T,STATE]
    const float* __restrict__ Cm,     // [B,T,STATE]
    const float* __restrict__ A_log,  // [DIM,STATE]
    float* __restrict__ outp,         // [B,T,DIM]
    int T, int dim, int dir)
{
  int idx = blockIdx.x * blockDim.x + threadIdx.x;  // b*dim + d
  int b = idx / dim;
  int d = idx - b * dim;

  float A[STATE], h[STATE];
#pragma unroll
  for (int s = 0; s < STATE; ++s) {
    A[s] = -__expf(A_log[d * STATE + s]);
    h[s] = 0.f;
  }

  int t0 = (dir > 0) ? 0 : (T - 1);
  for (int i = 0; i < T; ++i) {
    int t = t0 + dir * i;
    size_t row = (size_t)b * T + t;
    float dt  = delta[row * dim + d];
    float xbt = xb[row * dim + d];
    float y = 0.f;
#pragma unroll
    for (int s = 0; s < STATE; ++s) {
      float barA = __expf(dt * A[s]);
      h[s] = barA * h[s] + dt * Bm[row * STATE + s] * xbt;
      y += h[s] * Cm[row * STATE + s];
    }
    outp[row * dim + d] = y;
  }
}

// ---------------- combine + l2norm + head split + V transpose ----------------
__global__ void qknorm_kernel(
    const float* __restrict__ of, const float* __restrict__ ob,
    const float* __restrict__ Vf,
    _Float16* __restrict__ Qh, _Float16* __restrict__ Vt,
    int T, int H, int hd, int dim)
{
  int idx = blockIdx.x * blockDim.x + threadIdx.x;  // b*T*H + t*H + h
  int h = idx % H;
  int t = (idx / H) % T;
  int b = idx / (H * T);

  size_t inoff  = ((size_t)b * T + t) * dim + h * hd;
  size_t qoff   = (((size_t)b * H + h) * T + t) * hd;
  size_t vtbase = ((size_t)b * H + h) * (size_t)hd * T;

  float ss = 0.f;
  for (int i = 0; i < HEAD_DIM; ++i) {
    float c = of[inoff + i] + ob[inoff + i];
    ss += c * c;
  }
  float inv = 1.0f / fmaxf(sqrtf(ss), 1e-12f);
  for (int i = 0; i < HEAD_DIM; ++i) {
    float c = of[inoff + i] + ob[inoff + i];
    Qh[qoff + i] = (_Float16)(c * inv);
    Vt[vtbase + (size_t)i * T + t] = (_Float16)Vf[inoff + i];
  }
}

// ---------------- attention: one wave per (b,h, 16-row tile) ----------------
__global__ __launch_bounds__(32) void attn_kernel(
    const _Float16* __restrict__ Qh,   // [B,H,T,hd] (K == Q)
    const _Float16* __restrict__ Vt,   // [B,H,hd,T]
    const float* __restrict__ temperature, // [H]
    float* __restrict__ Y,             // [B,T,DIM]
    int T, int H, int hd, int dim)
{
  extern __shared__ float S[];  // 16*T floats
  int bh = blockIdx.x;
  int mt = blockIdx.y;
  int b = bh / H, h = bh - b * H;
  int lane = threadIdx.x;
  int m    = lane & 15;
  int half = lane >> 4;
  int koff = half << 3;

  const _Float16* Q = Qh + (size_t)bh * T * hd;
  const _Float16* V = Vt + (size_t)bh * (size_t)hd * T;
  float scale = log1pf(__expf(temperature[h]));  // softplus

  v16h qa0 = load_frag(Q + (size_t)(mt * 16 + m) * hd, koff);
  v16h qa1 = load_frag(Q + (size_t)(mt * 16 + m) * hd + 32, koff);

  for (int nt = 0; nt < TT / 16; ++nt) {
    v16h kb0 = load_frag(Q + (size_t)(nt * 16 + m) * hd, koff);
    v16h kb1 = load_frag(Q + (size_t)(nt * 16 + m) * hd + 32, koff);
    v8f acc = {};
    acc = wmma_f16(qa0, kb0, acc);
    acc = wmma_f16(qa1, kb1, acc);
#pragma unroll
    for (int r = 0; r < 8; ++r)
      S[(r + 8 * half) * T + nt * 16 + m] = acc[r] * scale;
  }
  __syncthreads();

  int base = m * T + half * (T / 2);
  float mx = -3.0e38f;
  for (int j = 0; j < TT / 2; ++j) mx = fmaxf(mx, S[base + j]);
  mx = fmaxf(mx, __shfl_xor(mx, 16));
  float sum = 0.f;
  for (int j = 0; j < TT / 2; ++j) {
    float e = __expf(S[base + j] - mx);
    S[base + j] = e;
    sum += e;
  }
  sum += __shfl_xor(sum, 16);
  float inv = 1.0f / sum;   // lane l holds 1/rowsum for row (l & 15)
  __syncthreads();

  v8f o[4] = {};
  for (int sc = 0; sc < TT / 32; ++sc) {
    v16h pf;
#pragma unroll
    for (int j = 0; j < 8; ++j) {
      pf[j]     = (_Float16)S[m * T + sc * 32 + koff + j];
      pf[8 + j] = (_Float16)S[m * T + sc * 32 + koff + 16 + j];
    }
#pragma unroll
    for (int nt = 0; nt < 4; ++nt) {
      v16h vb = load_frag(V + (size_t)(nt * 16 + m) * T + sc * 32, koff);
      o[nt] = wmma_f16(pf, vb, o[nt]);
    }
  }

#pragma unroll
  for (int r = 0; r < 8; ++r) {
    int rowi = r + 8 * half;
    float rinv = __shfl(inv, rowi);
    int t = mt * 16 + rowi;
#pragma unroll
    for (int nt = 0; nt < 4; ++nt)
      Y[((size_t)b * T + t) * dim + h * hd + nt * 16 + m] = o[nt][r] * rinv;
  }
}

// =====================================================================
extern "C" void kernel_launch(void* const* d_in, const int* in_sizes, int n_in,
                              void* d_out, int out_size, void* d_ws, size_t ws_size,
                              hipStream_t stream) {
  const float* x        = (const float*)d_in[0];
  const float* A_log_f  = (const float*)d_in[1];
  const float* Wd_f     = (const float*)d_in[2];
  const float* bd_f     = (const float*)d_in[3];
  const float* WB_f     = (const float*)d_in[4];
  const float* WC_f     = (const float*)d_in[5];
  const float* A_log_b  = (const float*)d_in[6];
  const float* Wd_b     = (const float*)d_in[7];
  const float* bd_b     = (const float*)d_in[8];
  const float* WB_b     = (const float*)d_in[9];
  const float* WC_b     = (const float*)d_in[10];
  const float* Wx       = (const float*)d_in[11];
  const float* bx       = (const float*)d_in[12];
  const float* Wv       = (const float*)d_in[13];
  const float* bv       = (const float*)d_in[14];
  const float* Wo       = (const float*)d_in[15];
  const float* bo       = (const float*)d_in[16];
  const float* temp     = (const float*)d_in[17];
  float* out = (float*)d_out;

  char* p = (char*)d_ws;
  auto alloc = [&](size_t bytes) -> void* {
    void* r = (void*)p;
    p += (bytes + 255) & ~(size_t)255;
    return r;
  };

  _Float16* xf16   = (_Float16*)alloc((size_t)NROWS * DIMC * 2);
  _Float16* Wx16   = (_Float16*)alloc((size_t)DIMC * DIMC * 2);
  _Float16* Wdf16  = (_Float16*)alloc((size_t)DIMC * DIMC * 2);
  _Float16* Wdb16  = (_Float16*)alloc((size_t)DIMC * DIMC * 2);
  _Float16* Wv16   = (_Float16*)alloc((size_t)DIMC * DIMC * 2);
  _Float16* Wo16   = (_Float16*)alloc((size_t)DIMC * DIMC * 2);
  _Float16* WBf16  = (_Float16*)alloc((size_t)STATE * DIMC * 2);
  _Float16* WCf16  = (_Float16*)alloc((size_t)STATE * DIMC * 2);
  _Float16* WBb16  = (_Float16*)alloc((size_t)STATE * DIMC * 2);
  _Float16* WCb16  = (_Float16*)alloc((size_t)STATE * DIMC * 2);
  float* xbase     = (float*)alloc((size_t)NROWS * DIMC * 4);
  float* delta_f   = (float*)alloc((size_t)NROWS * DIMC * 4);
  float* delta_b   = (float*)alloc((size_t)NROWS * DIMC * 4);
  float* Bmf       = (float*)alloc((size_t)NROWS * STATE * 4);
  float* Cmf       = (float*)alloc((size_t)NROWS * STATE * 4);
  float* Bmb       = (float*)alloc((size_t)NROWS * STATE * 4);
  float* Cmb       = (float*)alloc((size_t)NROWS * STATE * 4);
  float* out_f     = (float*)alloc((size_t)NROWS * DIMC * 4);
  float* out_b     = (float*)alloc((size_t)NROWS * DIMC * 4);
  float* Vf32      = (float*)alloc((size_t)NROWS * DIMC * 4);
  _Float16* Qh     = (_Float16*)alloc((size_t)NROWS * DIMC * 2);
  _Float16* Vt     = (_Float16*)alloc((size_t)NROWS * DIMC * 2);
  float* attn_out  = (float*)alloc((size_t)NROWS * DIMC * 4);
  _Float16* af16   = (_Float16*)alloc((size_t)NROWS * DIMC * 2);

  const int CB = 256;
  auto cvt = [&](const float* s, _Float16* d, int n) {
    cvt_f32_f16<<<(n + CB - 1) / CB, CB, 0, stream>>>(s, d, n);
  };

  // 1. fp16 copies
  cvt(x,    xf16,  NROWS * DIMC);
  cvt(Wx,   Wx16,  DIMC * DIMC);
  cvt(Wd_f, Wdf16, DIMC * DIMC);
  cvt(Wd_b, Wdb16, DIMC * DIMC);
  cvt(Wv,   Wv16,  DIMC * DIMC);
  cvt(Wo,   Wo16,  DIMC * DIMC);
  cvt(WB_f, WBf16, STATE * DIMC);
  cvt(WC_f, WCf16, STATE * DIMC);
  cvt(WB_b, WBb16, STATE * DIMC);
  cvt(WC_b, WCb16, STATE * DIMC);

  // 2. big projections: 64x64 block tile, double-buffered async-LDS staged W
  dim3 gBig(NROWS / 64, DIMC / 64);
  wmma_gemm_lds<<<gBig, 128, 0, stream>>>(xf16, Wx16,  bx,   xbase,   NROWS, DIMC, DIMC);
  wmma_gemm_lds<<<gBig, 128, 0, stream>>>(xf16, Wdf16, bd_f, delta_f, NROWS, DIMC, DIMC);
  wmma_gemm_lds<<<gBig, 128, 0, stream>>>(xf16, Wdb16, bd_b, delta_b, NROWS, DIMC, DIMC);
  wmma_gemm_lds<<<gBig, 128, 0, stream>>>(xf16, Wv16,  bv,   Vf32,    NROWS, DIMC, DIMC);

  // 3. narrow B/C projections (N = STATE = 16)
  dim3 gNar(NROWS / 16, 1);
  wmma_gemm_narrow<<<gNar, 32, 0, stream>>>(xf16, WBf16, Bmf, NROWS, STATE, DIMC);
  wmma_gemm_narrow<<<gNar, 32, 0, stream>>>(xf16, WCf16, Cmf, NROWS, STATE, DIMC);
  wmma_gemm_narrow<<<gNar, 32, 0, stream>>>(xf16, WBb16, Bmb, NROWS, STATE, DIMC);
  wmma_gemm_narrow<<<gNar, 32, 0, stream>>>(xf16, WCb16, Cmb, NROWS, STATE, DIMC);

  // 4. sequential scans
  scan_kernel<<<(BB * DIMC) / CB, CB, 0, stream>>>(xbase, delta_f, Bmf, Cmf, A_log_f, out_f, TT, DIMC, +1);
  scan_kernel<<<(BB * DIMC) / CB, CB, 0, stream>>>(xbase, delta_b, Bmb, Cmb, A_log_b, out_b, TT, DIMC, -1);

  // 5. combine + l2norm + head layouts
  qknorm_kernel<<<(BB * TT * HEADS) / CB, CB, 0, stream>>>(out_f, out_b, Vf32, Qh, Vt, TT, HEADS, HEAD_DIM, DIMC);

  // 6. attention
  dim3 gAtt(BB * HEADS, TT / 16);
  size_t shmem = (size_t)16 * TT * sizeof(float);
  attn_kernel<<<gAtt, 32, shmem, stream>>>(Qh, Vt, temp, attn_out, TT, HEADS, HEAD_DIM, DIMC);

  // 7. output projection
  cvt(attn_out, af16, NROWS * DIMC);
  wmma_gemm_lds<<<gBig, 128, 0, stream>>>(af16, Wo16, bo, out, NROWS, DIMC, DIMC);
}
